// CausalGNN_35811437314553
// MI455X (gfx1250) — compile-verified, matrix operand order checked by reading, not run on hardware
//
#include <hip/hip_runtime.h>
#include <hip/hip_bf16.h>
#include <math.h>

typedef __attribute__((ext_vector_type(16))) __bf16 v16bf;
typedef __attribute__((ext_vector_type(8)))  float  v8f;
typedef unsigned int u32x4 __attribute__((ext_vector_type(4)));
typedef int          i32x8 __attribute__((ext_vector_type(8)));
typedef int          i32x4 __attribute__((ext_vector_type(4)));

#define B_GR   512
#define S_SUB  65536
#define D_EMB  256
#define T_TASK 10
#define H2     512           // 2*D
#define THRESH 0.4f

// d_out (float): [0,5120) logits, [5120] loss, [5121, 5121+S) subgraph_mask
#define OUT_LOSS (B_GR * T_TASK)
#define OUT_MASK (B_GR * T_TASK + 1)

// d_ws (bytes): [0,262144) W1 swizzled bf16 (32 ctiles * 8 ksteps * 32 lanes * 16)
//               [262144, 327680) pass bytes [S]
//               [327680, 327684) cosine accumulator (float)
#define WS_W1B  0
#define WS_PASS 262144
#define WS_COS  327680

#define STAGE_ELEMS (4 * 8 * 32 * 16)   // 4 coltiles per stage = 16384 bf16 = 32KB

// ---------------------------------------------------------------------------
// Kernel 1: swizzle W1 (f32 -> bf16) into per-lane WMMA B-operand layout.
// B element e of lane L (n = L%16, hi = L/16): k = ks*32 + hi*8 + (e<8 ? e : e+8)
// Also zero the cosine accumulator (graph-replay determinism).
// ---------------------------------------------------------------------------
__global__ void cg_prep_kernel(const float* __restrict__ W1,
                               __bf16* __restrict__ w1b,
                               float* __restrict__ cosacc) {
    int idx = blockIdx.x * blockDim.x + threadIdx.x;
    if (idx == 0) *cosacc = 0.0f;
    if (idx >= 32 * 8 * 32 * 16) return;
    int e    = idx & 15;
    int lane = (idx >> 4) & 31;
    int ks   = (idx >> 9) & 7;
    int c    = idx >> 12;
    int n  = c * 16 + (lane & 15);
    int hi = lane >> 4;
    int k  = ks * 32 + hi * 8 + ((e < 8) ? e : (e + 8));
    w1b[idx] = (__bf16)W1[k * H2 + n];
}

// ---------------------------------------------------------------------------
// TDM: DMA one 32KB contiguous stage of swizzled W1 into LDS.
// D# group0: count=1, lds_addr, global_addr, type=2 (image)
// D# group1: data_size=8B, tensor_dim0=tile_dim0=4096, dim0_stride=4096, 1-D tile
// ---------------------------------------------------------------------------
__device__ __forceinline__ void cg_tdm_load_stage(const __bf16* gsrc, unsigned lds_off) {
    unsigned long long ga = (unsigned long long)(const void*)gsrc;
    u32x4 g0;
    g0[0] = 1u;                                               // count=1 (valid user D#)
    g0[1] = lds_off;                                          // LDS byte address
    g0[2] = (unsigned)(ga & 0xFFFFFFFFu);                     // global_addr[31:0]
    g0[3] = (unsigned)((ga >> 32) & 0x1FFFFFFu) | (2u << 30); // addr[56:32] | type=2
    i32x8 g1;
    g1[0] = (3 << 16);            // workgroup_mask=0, data_size=3 (8 bytes)
    g1[1] = (int)(4096u << 16);   // tensor_dim0 = 4096 (bits 79:48, low half)
    g1[2] = (1 << 16);            // tensor_dim0 hi=0, tensor_dim1 = 1
    g1[3] = (int)(4096u << 16);   // tile_dim0 = 4096 (bits 127:112)
    g1[4] = 0;                    // tile_dim1/tile_dim2 unused
    g1[5] = 4096;                 // tensor_dim0_stride (low 32)
    g1[6] = 0;
    g1[7] = 0;
    i32x4 z4 = {0, 0, 0, 0};
#if __clang_major__ >= 23
    i32x8 z8 = {0, 0, 0, 0, 0, 0, 0, 0};
    __builtin_amdgcn_tensor_load_to_lds(g0, g1, z4, z4, z8, 0);
#else
    __builtin_amdgcn_tensor_load_to_lds(g0, g1, z4, z4, 0);
#endif
}

// ---------------------------------------------------------------------------
// Kernel 2: gate MLP. Per 16-row tile of h_sub (one tile per wave, 8 waves/block):
//   H = ReLU(A @ W1 + b1)  via v_wmma_f32_16x16x32_bf16
//   B operand staged through LDS by the Tensor Data Mover, double-buffered:
//   8 stages x (4 coltiles x 8 ksteps) WMMAs per stage, fed by ds_load_b128.
//   logit = H @ W2 + b2 fused in epilogue; sigmoid -> d_out mask + pass bytes.
// ---------------------------------------------------------------------------
__global__ void __launch_bounds__(256)
cg_gate_kernel(const float* __restrict__ h_sub,
               const __bf16* __restrict__ w1b,
               const float* __restrict__ b1,
               const float* __restrict__ W2,
               const float* __restrict__ b2,
               float* __restrict__ out_mask,
               unsigned char* __restrict__ pass) {
    __shared__ __align__(128) __bf16 sB[2][STAGE_ELEMS];   // 2 x 32KB double buffer

    const int lane = threadIdx.x & 31;
    const int warp = threadIdx.x >> 5;
    const int m    = lane & 15;   // row (A) / col (B,C) within 16
    const int hi   = lane >> 4;
    const float b2v = b2[0];

    // 4096 row tiles, 128 blocks x 8 waves -> 4 uniform outer iterations.
    for (int t0 = blockIdx.x * 8; t0 < S_SUB / 16; t0 += 128 * 8) {
        const int tile = t0 + warp;

        // Build A operand: 16x256 f32 -> bf16 in WMMA 16-bit A layout.
        const float* arow = h_sub + (size_t)(tile * 16 + m) * D_EMB;
        v16bf a[8];
#pragma unroll
        for (int ks = 0; ks < 8; ++ks) {
            int k1 = ks * 32 + hi * 8;
            float4 f0 = *(const float4*)(arow + k1);
            float4 f1 = *(const float4*)(arow + k1 + 4);
            float4 f2 = *(const float4*)(arow + k1 + 16);
            float4 f3 = *(const float4*)(arow + k1 + 20);
            v16bf t;
            t[0]  = (__bf16)f0.x; t[1]  = (__bf16)f0.y; t[2]  = (__bf16)f0.z; t[3]  = (__bf16)f0.w;
            t[4]  = (__bf16)f1.x; t[5]  = (__bf16)f1.y; t[6]  = (__bf16)f1.z; t[7]  = (__bf16)f1.w;
            t[8]  = (__bf16)f2.x; t[9]  = (__bf16)f2.y; t[10] = (__bf16)f2.z; t[11] = (__bf16)f2.w;
            t[12] = (__bf16)f3.x; t[13] = (__bf16)f3.y; t[14] = (__bf16)f3.z; t[15] = (__bf16)f3.w;
            a[ks] = t;
        }

        float logit[8];
#pragma unroll
        for (int r = 0; r < 8; ++r) logit[r] = 0.0f;

        // Prologue: TDM stage 0 into buffer 0 (wave-level op, EXEC-independent).
        if (warp == 0)
            cg_tdm_load_stage(w1b, (unsigned)(unsigned long long)&sB[0][0]);

#pragma unroll 2
        for (int cg = 0; cg < 8; ++cg) {
            if (warp == 0) __builtin_amdgcn_s_wait_tensorcnt(0);
            __syncthreads();                       // stage cg visible to all waves
            if (warp == 0 && cg + 1 < 8)           // prefetch next stage (other buffer)
                cg_tdm_load_stage(w1b + (size_t)(cg + 1) * STAGE_ELEMS,
                                  (unsigned)(unsigned long long)&sB[(cg + 1) & 1][0]);

#pragma unroll
            for (int cl = 0; cl < 4; ++cl) {
                const int c = (cg << 2) + cl;
                v8f acc = {};
#pragma unroll
                for (int ks = 0; ks < 8; ++ks) {
                    v16bf bmat = *(const v16bf*)&sB[cg & 1][(((cl << 3) + ks) << 9) + (lane << 4)];
                    acc = __builtin_amdgcn_wmma_f32_16x16x32_bf16(
                        false, a[ks], false, bmat, (short)0, acc, false, false);
                }
                // epilogue for this 16-col tile: +b1, ReLU, * W2 column chunk
                float b1v = b1[c * 16 + m];
                float w2v = W2[c * 16 + m];
#pragma unroll
                for (int r = 0; r < 8; ++r) {
                    float h = acc[r] + b1v;        // C layout: row = hi*8 + r, col = m
                    h = h > 0.0f ? h : 0.0f;
                    logit[r] += h * w2v;
                }
            }
            __syncthreads();                       // all reads of this buffer done
        }

        // reduce the n-dimension partials across the 16 lanes of each half
#pragma unroll
        for (int r = 0; r < 8; ++r) {
            float v = logit[r];
            v += __shfl_xor(v, 1, 16);
            v += __shfl_xor(v, 2, 16);
            v += __shfl_xor(v, 4, 16);
            v += __shfl_xor(v, 8, 16);
            logit[r] = v;
        }
        if (m == 0) {
            int rbase = tile * 16 + hi * 8;
#pragma unroll
            for (int r = 0; r < 8; ++r) {
                float sg = 1.0f / (1.0f + __expf(-(logit[r] + b2v)));
                out_mask[rbase + r] = sg;
                pass[rbase + r]     = (sg > THRESH) ? 1 : 0;
            }
        }
    }
}

// ---------------------------------------------------------------------------
// Kernel 3: per-graph sparse masked mean + cosine + classifier.
// ---------------------------------------------------------------------------
#define LIST_MAX 2048

__device__ __forceinline__ float cg_block_sum(float v, float* red, int tid) {
    red[tid] = v;
    __syncthreads();
    for (int s = 128; s > 0; s >>= 1) {
        if (tid < s) red[tid] += red[tid + s];
        __syncthreads();
    }
    float r = red[0];
    __syncthreads();
    return r;
}

__global__ void cg_agg_kernel(const float* __restrict__ h_graph,
                              const float* __restrict__ h_sub,
                              const float* __restrict__ Wc,
                              const float* __restrict__ bc,
                              const unsigned char* __restrict__ mask,
                              const unsigned char* __restrict__ pass,
                              float* __restrict__ cosacc,
                              float* __restrict__ out_logits) {
    __shared__ int   s_list[LIST_MAX];
    __shared__ int   s_cnt;
    __shared__ float s_red[256];
    __shared__ float s_comb[H2];

    const int b   = blockIdx.x;
    const int tid = threadIdx.x;
    if (tid == 0) s_cnt = 0;
    __syncthreads();

    // Phase 1: scan this graph's mask row against the gate pass bits.
    const unsigned char* mrow = mask + (size_t)b * S_SUB;
    for (int j = tid; j < S_SUB; j += 256) {
        if (mrow[j] && pass[j]) {
            int p = atomicAdd(&s_cnt, 1);
            if (p < LIST_MAX) s_list[p] = j;
        }
    }
    __syncthreads();
    int cnt = s_cnt < LIST_MAX ? s_cnt : LIST_MAX;

    // Phase 2: cooperative gather-mean (thread tid owns dim tid).
    float acc = 0.0f;
    for (int e = 0; e < cnt; ++e)
        acc += h_sub[(size_t)s_list[e] * D_EMB + tid];
    float aligned = (cnt > 0) ? (acc / (float)cnt) : 0.0f;
    float g       = h_graph[(size_t)b * D_EMB + tid];
    s_comb[tid]         = g;
    s_comb[D_EMB + tid] = aligned;

    // Cosine of (aligned, g): dot / (||a|| * ||g||), guarded like the reference.
    float dsum  = cg_block_sum(aligned * g, s_red, tid);
    float nasum = cg_block_sum(aligned * aligned, s_red, tid);
    float ngsum = cg_block_sum(g * g, s_red, tid);
    if (tid == 0) {
        float na = sqrtf(nasum), ng = sqrtf(ngsum);
        float cv = dsum / (fmaxf(na, 1e-8f) * fmaxf(ng, 1e-8f));
        atomicAdd(cosacc, cv);
    }

    // Classifier: logits[b, t] = [g ; aligned] . Wc[:, t] + bc[t]
    if (tid < T_TASK) {
        float s = bc[tid];
        for (int k = 0; k < H2; ++k) s += s_comb[k] * Wc[k * T_TASK + tid];
        out_logits[b * T_TASK + tid] = s;
    }
}

__global__ void cg_fin_kernel(const float* __restrict__ cosacc,
                              float* __restrict__ out_loss) {
    if (blockIdx.x == 0 && threadIdx.x == 0)
        out_loss[0] = 1.0f - cosacc[0] / (float)B_GR;
}

// ---------------------------------------------------------------------------
extern "C" void kernel_launch(void* const* d_in, const int* in_sizes, int n_in,
                              void* d_out, int out_size, void* d_ws, size_t ws_size,
                              hipStream_t stream) {
    const float* h_graph = (const float*)d_in[0];
    const float* h_sub   = (const float*)d_in[1];
    const float* W1      = (const float*)d_in[2];
    const float* b1      = (const float*)d_in[3];
    const float* W2      = (const float*)d_in[4];
    const float* b2      = (const float*)d_in[5];
    const float* Wc      = (const float*)d_in[6];
    const float* bc      = (const float*)d_in[7];
    const unsigned char* mask = (const unsigned char*)d_in[8];

    char* ws = (char*)d_ws;
    __bf16*        w1b    = (__bf16*)(ws + WS_W1B);
    unsigned char* pass   = (unsigned char*)(ws + WS_PASS);
    float*         cosacc = (float*)(ws + WS_COS);

    float* out        = (float*)d_out;
    float* out_logits = out;
    float* out_loss   = out + OUT_LOSS;
    float* out_mask   = out + OUT_MASK;

    cg_prep_kernel<<<512, 256, 0, stream>>>(W1, w1b, cosacc);
    cg_gate_kernel<<<128, 256, 0, stream>>>(h_sub, w1b, b1, W2, b2, out_mask, pass);
    cg_agg_kernel<<<B_GR, 256, 0, stream>>>(h_graph, h_sub, Wc, bc, mask, pass,
                                            cosacc, out_logits);
    cg_fin_kernel<<<1, 32, 0, stream>>>(cosacc, out_loss);
}
